// BasicNMPBlock_28312424415493
// MI455X (gfx1250) — compile-verified, matrix operand order checked by reading, not run on hardware
//
#include <hip/hip_runtime.h>

typedef __attribute__((ext_vector_type(16))) _Float16 v16h;
typedef __attribute__((ext_vector_type(8)))  _Float16 half8;
typedef __attribute__((ext_vector_type(8)))  float    v8f;

#define BS  16
#define NN  2048
#define DIM 128

__device__ __forceinline__ float fast_sigmoid(float x) {
    return 1.0f / (1.0f + __expf(-x));
}
__device__ __forceinline__ float fast_tanh(float x) {
    // tanh(x) = 1 - 2/(exp(2x)+1); saturates correctly at +-1 for large |x|
    return 1.0f - 2.0f / (1.0f + __expf(2.0f * x));
}
__device__ __forceinline__ v16h cat16(half8 lo, half8 hi8) {
    return __builtin_shufflevector(lo, hi8, 0,1,2,3,4,5,6,7,8,9,10,11,12,13,14,15);
}
__device__ __forceinline__ half8 cvt8(const float* __restrict__ p) {
    const float4 f0 = *(const float4*)p;
    const float4 f1 = *(const float4*)(p + 4);
    half8 h;
    h[0]=(_Float16)f0.x; h[1]=(_Float16)f0.y; h[2]=(_Float16)f0.z; h[3]=(_Float16)f0.w;
    h[4]=(_Float16)f1.x; h[5]=(_Float16)f1.y; h[6]=(_Float16)f1.z; h[7]=(_Float16)f1.w;
    return h;
}

// -------- prep: spatial embedding, AoS float4 {sx, sy, sz, |s|^2} per node ----
__global__ __launch_bounds__(256) void prep_s_kernel(
    const float* __restrict__ x, const float* __restrict__ Ws,
    const float* __restrict__ bs, float4* __restrict__ s4)
{
    int idx = blockIdx.x * blockDim.x + threadIdx.x;
    if (idx >= BS * NN) return;
    const float* xr = x + (size_t)idx * DIM;
    float s0 = 0.f, s1 = 0.f, s2 = 0.f;
    #pragma unroll 4
    for (int k = 0; k < DIM; ++k) {
        float xv = xr[k];
        s0 += Ws[k] * xv;
        s1 += Ws[DIM + k] * xv;
        s2 += Ws[2 * DIM + k] * xv;
    }
    s0 += bs[0]; s1 += bs[1]; s2 += bs[2];
    s4[idx] = make_float4(s0, s1, s2, s0 * s0 + s1 * s1 + s2 * s2);
}

// -------- prep: f16 copies of Wm / Wi / Wh --------
__global__ __launch_bounds__(256) void prep_w_kernel(
    const float* __restrict__ Wm, const float* __restrict__ Wi,
    const float* __restrict__ Wh, _Float16* __restrict__ WmH,
    _Float16* __restrict__ WiH, _Float16* __restrict__ WhH)
{
    int idx = blockIdx.x * blockDim.x + threadIdx.x;
    if (idx < 3 * DIM * DIM) {
        WiH[idx] = (_Float16)Wi[idx];
        WhH[idx] = (_Float16)Wh[idx];
        if (idx < DIM * DIM) WmH[idx] = (_Float16)Wm[idx];
    }
}

// -------- prep: msg = relu(x @ Wm^T + bm) via WMMA, stored f16 TRANSPOSED ----
// msgT[b][d][i] so attention B-operand reads are contiguous along K (= i).
__global__ __launch_bounds__(128) void prep_msg_wmma(
    const float* __restrict__ x, const _Float16* __restrict__ WmH,
    const float* __restrict__ bm, _Float16* __restrict__ msgT)
{
    const int wave = threadIdx.x >> 5;
    const int lane = threadIdx.x & 31;
    const int l16  = lane & 15;
    const int hi   = lane >> 4;
    const int t    = blockIdx.x * 4 + wave;   // wave-tile id 0..2047
    const int b    = t >> 7;
    const int tb   = (t & 127) << 4;

    // A operand: 16 rows of x, K = 128 in 4 chunks of 32 (f32 -> f16)
    v16h a_x[4];
    const float* xr = x + ((size_t)b * NN + tb + l16) * DIM;
    #pragma unroll
    for (int c = 0; c < 4; ++c) {
        const int k0 = c * 32 + 8 * hi;
        a_x[c] = cat16(cvt8(xr + k0), cvt8(xr + k0 + 16));
    }

    #pragma unroll
    for (int s = 0; s < 8; ++s) {
        const _Float16* w = WmH + (size_t)(s * 16 + l16) * DIM;
        half8 lo[4], hi8[4];
        #pragma unroll
        for (int c = 0; c < 4; ++c) {         // batched load clause
            const int k0 = c * 32 + 8 * hi;
            lo[c]  = *(const half8*)(w + k0);
            hi8[c] = *(const half8*)(w + k0 + 16);
        }
        v8f acc = {};
        #pragma unroll
        for (int c = 0; c < 4; ++c)
            acc = __builtin_amdgcn_wmma_f32_16x16x32_f16(
                false, a_x[c], false, cat16(lo[c], hi8[c]), (short)0, acc, false, false);
        const int col  = s * 16 + l16;        // output feature d
        const float bv = bm[col];
        #pragma unroll
        for (int r = 0; r < 8; ++r) {
            const int row = tb + r + 8 * hi;  // node index i (contiguous over r)
            msgT[((size_t)b * DIM + col) * NN + row] =
                (_Float16)fmaxf(acc[r] + bv, 0.0f);
        }
    }
}

// =======================================================================
// Fused attention: A(i,j) = exp(-(sq_i+sq_j-2 s_i.s_j)) * mask, x_nmp = A @ msg
// Block = 128 threads (4 waves); wave owns 16 rows x 128 cols of output.
// The whole batch's s/sq table (32 KB) is staged in LDS ONCE, so the main
// loop is barrier-free: issue 16 B loads, overlap A-tile math (exp) with the
// load latency, then 8 back-to-back WMMAs per 32-wide K step.
// =======================================================================
__global__ __launch_bounds__(128) void attn_kernel(
    const float* __restrict__ mask,
    const float4* __restrict__ s4,
    const _Float16* __restrict__ msgT,
    _Float16* __restrict__ xnmp)
{
    __shared__ float4 ls4[NN];           // 32 KB: s/sq for ALL j of this batch
    const int b    = blockIdx.x >> 5;    // 16 batches
    const int tile = blockIdx.x & 31;    // 32 row-tiles of 64
    const int wave = threadIdx.x >> 5;
    const int lane = threadIdx.x & 31;
    const int l16  = lane & 15;
    const int hi   = lane >> 4;
    const int i    = tile * 64 + wave * 16 + l16;    // this lane's A row

    // stage entire s4 table for this batch (coalesced b128 copies), one barrier
    const float4* s4b = s4 + b * NN;
    for (int q = threadIdx.x; q < NN; q += 128) ls4[q] = s4b[q];
    __syncthreads();

    const float4 sv  = ls4[i];
    const float  six = sv.x, siy = sv.y, siz = sv.z, sqi = sv.w;

    const float*    mrow = mask + ((size_t)b * NN + i) * NN;
    const _Float16* mT   = msgT + (size_t)b * DIM * NN;

    v8f acc[8] = {};

    for (int m = 0; m < NN; m += 32) {
        // 1) issue all B loads for this K step (16 x b128, one clause)
        half8 blo[8], bhi8[8];
        #pragma unroll
        for (int s = 0; s < 8; ++s) {
            const _Float16* bp = mT + (size_t)(s * 16 + l16) * NN + m + 8 * hi;
            blo[s]  = *(const half8*)bp;
            bhi8[s] = *(const half8*)(bp + 16);
        }

        // 2) build A tile (16x32 f16) in WMMA register layout while B flies:
        //    lane<16: K in {0..7,16..23}; lane>=16: K in {8..15,24..31}
        v16h a;
        #pragma unroll
        for (int g = 0; g < 2; ++g) {
            const int jb = g * 16 + 8 * hi;
            const float4 mk0 = *(const float4*)(mrow + m + jb);
            const float4 mk1 = *(const float4*)(mrow + m + jb + 4);
            const float mk[8] = {mk0.x, mk0.y, mk0.z, mk0.w, mk1.x, mk1.y, mk1.z, mk1.w};
            #pragma unroll
            for (int e = 0; e < 8; ++e) {
                const float4 sj = ls4[m + jb + e];
                const float d = sqi + sj.w - 2.0f * (six * sj.x + siy * sj.y + siz * sj.z);
                a[g * 8 + e] = (_Float16)(__expf(-d) * mk[e]);
            }
        }

        // 3) 8 WMMAs, one per 16-wide slice of the 128 output features
        #pragma unroll
        for (int s = 0; s < 8; ++s)
            acc[s] = __builtin_amdgcn_wmma_f32_16x16x32_f16(
                false, a, false, cat16(blo[s], bhi8[s]), (short)0, acc[s], false, false);
    }

    // C layout: VGPR r -> row M = r + 8*hi, lane16 -> column
    #pragma unroll
    for (int s = 0; s < 8; ++s) {
        #pragma unroll
        for (int r = 0; r < 8; ++r) {
            const int row = tile * 64 + wave * 16 + r + 8 * hi;
            const int col = s * 16 + l16;
            xnmp[((size_t)b * NN + row) * DIM + col] = (_Float16)acc[s][r];
        }
    }
}

// =======================================================================
// GRU update: gi = x_nmp @ Wi^T + bi, gh = x @ Wh^T + bh, gates, output.
// One wave per 16-row tile. Column group sg: r/z/n gates land at output cols
// sg*16, 128+sg*16, 256+sg*16 -> purely elementwise on matching v8f accums.
// =======================================================================
__global__ __launch_bounds__(128) void gru_kernel(
    const float* __restrict__ x,
    const _Float16* __restrict__ xnmp,
    const _Float16* __restrict__ WiH, const _Float16* __restrict__ WhH,
    const float* __restrict__ bi, const float* __restrict__ bh,
    float* __restrict__ out)
{
    const int wave = threadIdx.x >> 5;
    const int lane = threadIdx.x & 31;
    const int l16  = lane & 15;
    const int hi   = lane >> 4;
    const int t    = blockIdx.x * 4 + wave;   // wave-tile id 0..2047
    const int b    = t >> 7;
    const int tb   = (t & 127) << 4;          // row base within batch

    // A operands (16 rows x K=128), 4 k-chunks of 32
    v16h a_x[4], a_n[4];
    const float*    xr = x    + ((size_t)b * NN + tb + l16) * DIM;
    const _Float16* nr = xnmp + ((size_t)b * NN + tb + l16) * DIM;
    #pragma unroll
    for (int c = 0; c < 4; ++c) {
        const int k0 = c * 32 + 8 * hi;
        a_x[c] = cat16(cvt8(xr + k0), cvt8(xr + k0 + 16));
        a_n[c] = cat16(*(const half8*)(nr + k0), *(const half8*)(nr + k0 + 16));
    }

    for (int sg = 0; sg < 8; ++sg) {
        v8f ai[3] = {};
        v8f ah[3] = {};
        #pragma unroll
        for (int g = 0; g < 3; ++g) {             // r / z / n gate weight rows
            const int nout = g * DIM + sg * 16 + l16;
            const _Float16* wi = WiH + (size_t)nout * DIM;
            const _Float16* wh = WhH + (size_t)nout * DIM;
            half8 wlo[4], whi[4], ulo[4], uhi[4];
            #pragma unroll
            for (int c = 0; c < 4; ++c) {         // batched 16-load clause
                const int k0 = c * 32 + 8 * hi;
                wlo[c] = *(const half8*)(wi + k0);
                whi[c] = *(const half8*)(wi + k0 + 16);
                ulo[c] = *(const half8*)(wh + k0);
                uhi[c] = *(const half8*)(wh + k0 + 16);
            }
            #pragma unroll
            for (int c = 0; c < 4; ++c) {         // 8 back-to-back WMMAs
                ai[g] = __builtin_amdgcn_wmma_f32_16x16x32_f16(
                    false, a_n[c], false, cat16(wlo[c], whi[c]), (short)0, ai[g], false, false);
                ah[g] = __builtin_amdgcn_wmma_f32_16x16x32_f16(
                    false, a_x[c], false, cat16(ulo[c], uhi[c]), (short)0, ah[g], false, false);
            }
        }
        const int col = sg * 16 + l16;
        const float bir = bi[col],           bhr = bh[col];
        const float biz = bi[DIM + col],     bhz = bh[DIM + col];
        const float bin = bi[2 * DIM + col], bhn = bh[2 * DIM + col];
        #pragma unroll
        for (int r = 0; r < 8; ++r) {
            const int row = tb + r + 8 * hi;
            const float rg = fast_sigmoid(ai[0][r] + bir + ah[0][r] + bhr);
            const float zg = fast_sigmoid(ai[1][r] + biz + ah[1][r] + bhz);
            const float ng = fast_tanh((ai[2][r] + bin) + rg * (ah[2][r] + bhn));
            const float hv = x[((size_t)b * NN + row) * DIM + col];
            out[((size_t)b * NN + row) * DIM + col] = (1.0f - zg) * ng + zg * hv;
        }
    }
}

extern "C" void kernel_launch(void* const* d_in, const int* in_sizes, int n_in,
                              void* d_out, int out_size, void* d_ws, size_t ws_size,
                              hipStream_t stream)
{
    (void)in_sizes; (void)n_in; (void)out_size; (void)ws_size;
    const float* x    = (const float*)d_in[0];
    const float* mask = (const float*)d_in[1];
    const float* Ws   = (const float*)d_in[2];
    const float* bs   = (const float*)d_in[3];
    const float* Wm   = (const float*)d_in[4];
    const float* bm   = (const float*)d_in[5];
    const float* Wi   = (const float*)d_in[6];
    const float* bi   = (const float*)d_in[7];
    const float* Wh   = (const float*)d_in[8];
    const float* bh   = (const float*)d_in[9];
    float* out = (float*)d_out;

    // workspace layout (~17.6 MB)
    float4*   s4   = (float4*)d_ws;                   // [BS*NN] {sx,sy,sz,sq}
    _Float16* msgT = (_Float16*)(s4 + BS * NN);       // [BS][DIM][NN] f16
    _Float16* xnmp = msgT + (size_t)BS * DIM * NN;    // [BS][NN][DIM] f16
    _Float16* WiH  = xnmp + (size_t)BS * NN * DIM;    // [3*DIM][DIM] f16
    _Float16* WhH  = WiH + 3 * DIM * DIM;             // [3*DIM][DIM] f16
    _Float16* WmH  = WhH + 3 * DIM * DIM;             // [DIM][DIM] f16

    prep_s_kernel<<<(BS * NN + 255) / 256, 256, 0, stream>>>(x, Ws, bs, s4);
    prep_w_kernel<<<(3 * DIM * DIM + 255) / 256, 256, 0, stream>>>(Wm, Wi, Wh, WmH, WiH, WhH);
    prep_msg_wmma<<<BS * NN / 64, 128, 0, stream>>>(x, WmH, bm, msgT);
    attn_kernel<<<BS * 32, 128, 0, stream>>>(mask, s4, msgT, xnmp);
    gru_kernel<<<BS * NN / 64, 128, 0, stream>>>(x, xnmp, WiH, WhH, bi, bh, out);
}
